// AttentionMIL_15985868275832
// MI455X (gfx1250) — matrix operand order repaired and checked by reading, not compiled, not run on hardware
//
#include <hip/hip_runtime.h>
#include <hip/hip_bf16.h>

// ---------------------------------------------------------------------------
// AttentionMIL on MI455X (gfx1250, wave32, WMMA)
// scores = W2 . relu(W1 @ F + b1) + b2  as implicit GEMM via
// v_wmma_f32_16x16x32_f16 (f16 inputs, f32 accumulate), then masked softmax,
// bag accumulation and BCE-with-logits loss.
// ---------------------------------------------------------------------------

#define NB    32
#define NC    64
#define NHID  128
#define HWPX  65536            // 256*256
#define TILE_PX 256            // pixels per workgroup in score kernel
#define ROWB  144              // LDS row stride (bytes): 36 dwords, 16B aligned
#define NEG_FILL (-1e9f)

typedef __attribute__((ext_vector_type(16))) _Float16 v16h;
typedef __attribute__((ext_vector_type(8)))  float    v8f;

union HV16 { v16h v; uint4 q[2]; };

static __device__ __forceinline__ unsigned int pkh2(float a, float b) {
    union { _Float16 h; unsigned short u; } ca, cb;
    ca.h = (_Float16)a; cb.h = (_Float16)b;
    return (unsigned int)ca.u | ((unsigned int)cb.u << 16);
}
static __device__ __forceinline__ unsigned short f2h(float a) {
    union { _Float16 h; unsigned short u; } c;
    c.h = (_Float16)a; return c.u;
}

// ---------------------------------------------------------------------------
// Kernel 1: per-pixel MLP scores via WMMA. Writes masked score (or -1e9)
// into att[] (aliases d_out+1, later overwritten in-place with weights).
// ---------------------------------------------------------------------------
__global__ __launch_bounds__(128, 4) void mil_scores_kernel(
    const float* __restrict__ feat, const float* __restrict__ W1,
    const float* __restrict__ b1,   const float* __restrict__ W2,
    const float* __restrict__ b2,   const int* __restrict__ zone,
    const int* __restrict__ cats,   float* __restrict__ att)
{
    __shared__ __align__(16) unsigned char smF[TILE_PX * ROWB];   // 36864 B
    __shared__ __align__(16) unsigned char smW1[NHID * ROWB];     // 18432 B
    __shared__ __align__(16) float smB1[NHID];
    __shared__ __align__(16) float smW2[NHID];

    const int tid = threadIdx.x;
    const long long p0 = (long long)blockIdx.x * TILE_PX;   // global pixel base
    const int  b   = (int)(p0 >> 16);                       // HWPX == 1<<16
    const long long hw0 = p0 & (HWPX - 1);
    const float* fbase = feat + (long long)b * NC * HWPX;

    // ---- Stage feature tile: fp32 (C,hw) -> f16 pixel-major LDS rows ----
    // Each thread owns two adjacent pixels (float2 global loads, 256B/wave).
    {
        const int px0 = 2 * tid;                  // 0..254
        const long long hw = hw0 + px0;
        unsigned char* row0 = smF + px0 * ROWB;
        unsigned char* row1 = row0 + ROWB;
        #pragma unroll
        for (int c0 = 0; c0 < NC; c0 += 8) {
            float2 v[8];
            #pragma unroll
            for (int i = 0; i < 8; ++i)
                v[i] = *(const float2*)&fbase[(long long)(c0 + i) * HWPX + hw];
            uint4 q0, q1;
            q0.x = pkh2(v[0].x, v[1].x); q0.y = pkh2(v[2].x, v[3].x);
            q0.z = pkh2(v[4].x, v[5].x); q0.w = pkh2(v[6].x, v[7].x);
            q1.x = pkh2(v[0].y, v[1].y); q1.y = pkh2(v[2].y, v[3].y);
            q1.z = pkh2(v[4].y, v[5].y); q1.w = pkh2(v[6].y, v[7].y);
            *(uint4*)(row0 + c0 * 2) = q0;
            *(uint4*)(row1 + c0 * 2) = q1;
        }
    }
    // ---- Stage W1 as f16 (row stride ROWB), plus b1 / W2 ----
    for (int e = tid; e < NHID * NC; e += 128) {
        const int r = e >> 6, k = e & 63;
        *(unsigned short*)(smW1 + r * ROWB + k * 2) = f2h(W1[e]);
    }
    if (tid < NHID) { smB1[tid] = b1[tid]; smW2[tid] = W2[tid]; }
    __syncthreads();

    const int lane = tid & 31;
    const int wave = tid >> 5;            // 4 waves, 64 pixels each
    const int hi   = lane >> 4;           // lane half
    const int lo   = lane & 15;
    const int waveBase = wave * 64;

    // ---- Load B operands (F tile) once: 4 N-tiles x 2 K-steps ----
    // B 32x16 f16, this code's k order: lanes 0-15 col N=l hold K=0..15; lanes 16-31 same
    // col hold K=16..31 (consecutive pairs per VGPR).
    v16h Bop[8];
    #pragma unroll
    for (int nt = 0; nt < 4; ++nt) {
        const int prow = waveBase + nt * 16 + lo;
        #pragma unroll
        for (int ks = 0; ks < 2; ++ks) {
            const unsigned char* bp = smF + prow * ROWB + (ks * 32 + hi * 16) * 2;
            HV16 t;
            t.q[0] = *(const uint4*)bp;
            t.q[1] = *(const uint4*)(bp + 16);
            Bop[nt * 2 + ks] = t.v;
        }
    }

    float partial[4] = {0.f, 0.f, 0.f, 0.f};

    // Keep this a real loop: one live accumulator set -> <=256 VGPRs,
    // no vgpr-msb bank switching, better occupancy.
    #pragma unroll 1
    for (int mt = 0; mt < 8; ++mt) {
        v8f acc[4] = {};
        #pragma unroll
        for (int ks = 0; ks < 2; ++ks) {
            // A 16x32 f16, this code's k order: lane pair (l, l+16) both hold row M=l&15;
            // lane<16: K=0..15; lane>=16: K=16..31 (+ks*32).
            const unsigned char* ap =
                smW1 + (mt * 16 + lo) * ROWB + (ks * 32 + hi * 16) * 2;
            HV16 t;
            t.q[0] = *(const uint4*)ap;
            t.q[1] = *(const uint4*)(ap + 16);
            const v16h A = t.v;
            #pragma unroll
            for (int nt = 0; nt < 4; ++nt)
                acc[nt] = __builtin_amdgcn_wmma_f32_16x16x32_f16(
                    false, A, false, Bop[nt * 2 + ks], (short)0, acc[nt],
                    false, false);
        }
        // Epilogue: rows of this lane are hb..hb+7 (D-matrix VGPR layout)
        const int hb = mt * 16 + hi * 8;
        const float4 b1a = *(const float4*)(smB1 + hb);
        const float4 b1b = *(const float4*)(smB1 + hb + 4);
        const float4 w2a = *(const float4*)(smW2 + hb);
        const float4 w2b = *(const float4*)(smW2 + hb + 4);
        const float b1v[8] = {b1a.x, b1a.y, b1a.z, b1a.w,
                              b1b.x, b1b.y, b1b.z, b1b.w};
        const float w2v[8] = {w2a.x, w2a.y, w2a.z, w2a.w,
                              w2b.x, w2b.y, w2b.z, w2b.w};
        #pragma unroll
        for (int nt = 0; nt < 4; ++nt) {
            #pragma unroll
            for (int j = 0; j < 8; ++j) {
                float h = acc[nt][j] + b1v[j];
                h = fmaxf(h, 0.f);
                partial[nt] = fmaf(h, w2v[j], partial[nt]);
            }
        }
    }

    const float bias2 = b2[0];
    const int   cat   = cats[b];
    #pragma unroll
    for (int nt = 0; nt < 4; ++nt) {
        const float other = __shfl_xor(partial[nt], 16, 32); // combine row halves
        const float s = partial[nt] + other + bias2;
        if (lane < 16) {
            const long long p = p0 + waveBase + nt * 16 + lo;
            const int z = zone[p];
            att[p] = (z == cat && z > 0) ? s : NEG_FILL;
        }
    }
}

// ---------------------------------------------------------------------------
// Kernel 2a: per-batch max of att + count of valid (unmasked) pixels
// ---------------------------------------------------------------------------
__global__ __launch_bounds__(1024) void batch_max_kernel(
    const float* __restrict__ att, float* __restrict__ maxv,
    float* __restrict__ cntv)
{
    const int b = blockIdx.x;
    const float* a = att + (long long)b * HWPX;
    float mx = -3.0e38f, cnt = 0.f;
    for (int i = threadIdx.x; i < HWPX; i += 1024) {
        const float v = a[i];
        mx = fmaxf(mx, v);
        cnt += (v > -1.0e8f) ? 1.f : 0.f;
    }
    #pragma unroll
    for (int off = 16; off; off >>= 1) {
        mx = fmaxf(mx, __shfl_xor(mx, off, 32));
        cnt += __shfl_xor(cnt, off, 32);
    }
    __shared__ float smx[32], scn[32];
    if ((threadIdx.x & 31) == 0) { smx[threadIdx.x >> 5] = mx; scn[threadIdx.x >> 5] = cnt; }
    __syncthreads();
    if (threadIdx.x < 32) {
        mx = smx[threadIdx.x]; cnt = scn[threadIdx.x];
        #pragma unroll
        for (int off = 16; off; off >>= 1) {
            mx = fmaxf(mx, __shfl_xor(mx, off, 32));
            cnt += __shfl_xor(cnt, off, 32);
        }
        if (threadIdx.x == 0) { maxv[b] = mx; cntv[b] = cnt; }
    }
}

// ---------------------------------------------------------------------------
// Kernel 2b: per-batch sum of exp(att - max)
// ---------------------------------------------------------------------------
__global__ __launch_bounds__(1024) void batch_sum_kernel(
    const float* __restrict__ att, const float* __restrict__ maxv,
    float* __restrict__ sumv)
{
    const int b = blockIdx.x;
    const float mx = maxv[b];
    const float* a = att + (long long)b * HWPX;
    float s = 0.f;
    for (int i = threadIdx.x; i < HWPX; i += 1024)
        s += __expf(a[i] - mx);
    #pragma unroll
    for (int off = 16; off; off >>= 1) s += __shfl_xor(s, off, 32);
    __shared__ float ssm[32];
    if ((threadIdx.x & 31) == 0) ssm[threadIdx.x >> 5] = s;
    __syncthreads();
    if (threadIdx.x < 32) {
        s = ssm[threadIdx.x];
        #pragma unroll
        for (int off = 16; off; off >>= 1) s += __shfl_xor(s, off, 32);
        if (threadIdx.x == 0) sumv[b] = s;
    }
}

// ---------------------------------------------------------------------------
// Kernel 3: weights (softmax, zeroed if !has) written in-place over att;
// per-block bag partials (deterministic; one block == 256 pixels, one batch)
// ---------------------------------------------------------------------------
__global__ __launch_bounds__(256) void weights_bag_kernel(
    float* __restrict__ att, const float* __restrict__ logits,
    const int* __restrict__ zone, const int* __restrict__ cats,
    const float* __restrict__ maxv, const float* __restrict__ sumv,
    const float* __restrict__ cntv, float* __restrict__ partials)
{
    const long long p = (long long)blockIdx.x * 256 + threadIdx.x;
    const int b = (int)(p >> 16);
    const float mx = maxv[b], sm = sumv[b];
    const bool has = cntv[b] > 0.5f;
    const float v = att[p];
    const float w = __expf(v - mx) / sm;
    const float wout = has ? w : 0.f;
    att[p] = wout;                                    // final attention map
    const int z = zone[p];
    const bool mask = (z == cats[b]) && (z > 0);
    float contrib = mask ? logits[p] * wout : 0.f;
    #pragma unroll
    for (int off = 16; off; off >>= 1) contrib += __shfl_xor(contrib, off, 32);
    __shared__ float red[8];
    if ((threadIdx.x & 31) == 0) red[threadIdx.x >> 5] = contrib;
    __syncthreads();
    if (threadIdx.x == 0) {
        float s = 0.f;
        #pragma unroll
        for (int i = 0; i < 8; ++i) s += red[i];
        partials[blockIdx.x] = s;
    }
}

// ---------------------------------------------------------------------------
// Kernel 4: reduce 256 partials/batch -> bag, BCE-with-logits mean -> out[0]
// ---------------------------------------------------------------------------
__global__ __launch_bounds__(256) void finalize_kernel(
    const float* __restrict__ partials, const float* __restrict__ labels,
    float* __restrict__ out)
{
    __shared__ float red[8];
    __shared__ float total;
    if (threadIdx.x == 0) total = 0.f;
    __syncthreads();
    for (int b = 0; b < NB; ++b) {
        float v = partials[b * 256 + threadIdx.x];
        #pragma unroll
        for (int off = 16; off; off >>= 1) v += __shfl_xor(v, off, 32);
        if ((threadIdx.x & 31) == 0) red[threadIdx.x >> 5] = v;
        __syncthreads();
        if (threadIdx.x == 0) {
            float x = 0.f;
            #pragma unroll
            for (int i = 0; i < 8; ++i) x += red[i];
            const float y = labels[b];
            total += fmaxf(x, 0.f) - x * y + log1pf(__expf(-fabsf(x)));
        }
        __syncthreads();
    }
    if (threadIdx.x == 0) out[0] = total * (1.f / (float)NB);
}

// ---------------------------------------------------------------------------
extern "C" void kernel_launch(void* const* d_in, const int* in_sizes, int n_in,
                              void* d_out, int out_size, void* d_ws, size_t ws_size,
                              hipStream_t stream) {
    (void)in_sizes; (void)n_in; (void)out_size; (void)ws_size;
    const float* logits = (const float*)d_in[0];
    const float* feat   = (const float*)d_in[1];
    const int*   zone   = (const int*)d_in[2];
    const int*   cats   = (const int*)d_in[3];
    const float* labels = (const float*)d_in[4];
    const float* W1     = (const float*)d_in[5];
    const float* b1     = (const float*)d_in[6];
    const float* W2     = (const float*)d_in[7];
    const float* b2     = (const float*)d_in[8];

    float* out = (float*)d_out;
    float* att = out + 1;                         // attention_maps region
    float* ws  = (float*)d_ws;
    float* maxv     = ws;                         // [32]
    float* sumv     = ws + 32;                    // [32]
    float* cntv     = ws + 64;                    // [32]
    float* partials = ws + 128;                   // [8192]

    const int npix_blocks = (NB * HWPX) / TILE_PX;   // 8192

    mil_scores_kernel<<<npix_blocks, 128, 0, stream>>>(
        feat, W1, b1, W2, b2, zone, cats, att);
    batch_max_kernel<<<NB, 1024, 0, stream>>>(att, maxv, cntv);
    batch_sum_kernel<<<NB, 1024, 0, stream>>>(att, maxv, sumv);
    weights_bag_kernel<<<(NB * HWPX) / 256, 256, 0, stream>>>(
        att, logits, zone, cats, maxv, sumv, cntv, partials);
    finalize_kernel<<<1, 256, 0, stream>>>(partials, labels, out);
}
